// Classifier_22625887715977
// MI455X (gfx1250) — matrix operand order repaired and checked by reading, not compiled, not run on hardware
//
#include <hip/hip_runtime.h>
#include <hip/hip_bf16.h>
#include <math.h>

typedef __attribute__((ext_vector_type(16))) _Float16 v16h;
typedef __attribute__((ext_vector_type(8)))  float    v8f;

#define FD    1024
#define SEMD  300
#define SEMP  320   // SEM padded to multiple of 32 for WMMA K dim
#define NB    512
#define BS    32
#define NWAY  5
#define BTOT  160   // BS*NWAY
#define NQ    75

// ---------------------------------------------------------------------------
// small VALU kernels (negligible FLOPs)
// ---------------------------------------------------------------------------

__device__ __forceinline__ float leaky01(float x) { return x >= 0.f ? x : 0.1f * x; }
__device__ __forceinline__ float sigm(float x) { return 1.f / (1.f + expf(-x)); }

// h1[160,300] = leaky(ss @ Wm1 + bm1)
__global__ void k_mlp1(const float* __restrict__ ss, const float* __restrict__ Wm1,
                       const float* __restrict__ bm1, float* __restrict__ h1) {
    int t = blockIdx.x * blockDim.x + threadIdx.x;
    if (t >= BTOT * SEMD) return;
    int r = t / SEMD, n = t % SEMD;
    float s = bm1[n];
    const float* row = ss + (size_t)r * SEMD;
    for (int d = 0; d < SEMD; ++d) s += row[d] * Wm1[(size_t)d * SEMD + n];
    h1[t] = leaky01(s);
}

// smap[160,300] = h1 @ Wm2 + bm2
__global__ void k_mlp2(const float* __restrict__ h1, const float* __restrict__ Wm2,
                       const float* __restrict__ bm2, float* __restrict__ smap) {
    int t = blockIdx.x * blockDim.x + threadIdx.x;
    if (t >= BTOT * SEMD) return;
    int r = t / SEMD, n = t % SEMD;
    float s = bm2[n];
    const float* row = h1 + (size_t)r * SEMD;
    for (int d = 0; d < SEMD; ++d) s += row[d] * Wm2[(size_t)d * SEMD + n];
    smap[t] = s;
}

// avg[32,1324] = mean over 512 of concat(base_weights[b], base_seman[b])
__global__ void k_avg(const float* __restrict__ bw, const float* __restrict__ bsm,
                      float* __restrict__ avg) {
    int t = blockIdx.x * blockDim.x + threadIdx.x;
    const int CD = FD + SEMD;
    if (t >= BS * CD) return;
    int b = t / CD, d = t % CD;
    float s = 0.f;
    if (d < FD) {
        const float* p = bw + (size_t)b * NB * FD + d;
        for (int k = 0; k < NB; ++k) s += p[(size_t)k * FD];
    } else {
        const float* p = bsm + (size_t)b * NB * SEMD + (d - FD);
        for (int k = 0; k < NB; ++k) s += p[(size_t)k * SEMD];
    }
    avg[t] = s * (1.f / NB);
}

// gate_vis[32,1024], gate_sem[32,300] = sigmoid(avg @ W + b) + 1
__global__ void k_gates(const float* __restrict__ avg, const float* __restrict__ Wvis,
                        const float* __restrict__ bvis, const float* __restrict__ Wsem,
                        const float* __restrict__ bsem, float* __restrict__ gvis,
                        float* __restrict__ gsem) {
    int t = blockIdx.x * blockDim.x + threadIdx.x;
    const int CD = FD + SEMD;
    if (t >= BS * CD) return;
    int b = t / CD, d = t % CD;
    const float* a = avg + (size_t)b * CD;
    if (d < FD) {
        float s = bvis[d];
        for (int i = 0; i < CD; ++i) s += a[i] * Wvis[(size_t)i * FD + d];
        gvis[(size_t)b * FD + d] = sigm(s) + 1.f;
    } else {
        int n = d - FD;
        float s = bsem[n];
        for (int i = 0; i < CD; ++i) s += a[i] * Wsem[(size_t)i * SEMD + n];
        gsem[(size_t)b * SEMD + n] = sigm(s) + 1.f;
    }
}

// ---------------------------------------------------------------------------
// fragment packers: f32 -> f16 in exact WMMA per-lane fragment layout
// A 16x32 f16: lane = m%16 + 16*((k%16)/8); idx = (k/16 within tile)*8 + k%8
// B 32x16 f16: lane = n%16 + 16*((k%16)/8); idx = (k/16 within tile)*8 + k%8
// ---------------------------------------------------------------------------

__global__ void k_pack_a(const float* __restrict__ src, const float* __restrict__ gate,
                         _Float16* __restrict__ dst, int rows, int cols, int colsPad,
                         size_t srcStride, size_t gateStride, size_t dstStride) {
    int b = blockIdx.y;
    size_t total = (size_t)rows * colsPad;
    size_t t = (size_t)blockIdx.x * blockDim.x + threadIdx.x;
    if (t >= total) return;
    int m = (int)(t / colsPad), k = (int)(t % colsPad);
    float v = 0.f;
    if (k < cols) {
        v = src[(size_t)b * srcStride + (size_t)m * cols + k];
        if (gate) v *= gate[(size_t)b * gateStride + k];
    }
    int mt = m >> 4, kt = k >> 5;
    int mp = m & 15, kp = k & 31;
    int lane = mp + 16 * ((kp & 15) >> 3);
    int idx  = ((kp >> 4) << 3) + (kp & 7);
    int nkt = colsPad >> 5;
    dst[(size_t)b * dstStride + (((size_t)mt * nkt + kt) * 32 + lane) * 16 + idx] = (_Float16)v;
}

__global__ void k_pack_b(const float* __restrict__ W, _Float16* __restrict__ dst,
                         int K, int Kpad, int N) {
    size_t total = (size_t)Kpad * N;
    size_t t = (size_t)blockIdx.x * blockDim.x + threadIdx.x;
    if (t >= total) return;
    int k = (int)(t / N), n = (int)(t % N);
    float v = (k < K) ? W[(size_t)k * N + n] : 0.f;
    int kt = k >> 5, nt = n >> 4;
    int kp = k & 31, np = n & 15;
    int lane = np + 16 * ((kp & 15) >> 3);
    int idx  = ((kp >> 4) << 3) + (kp & 7);
    int nkt = Kpad >> 5;
    dst[(((size_t)nt * nkt + kt) * 32 + lane) * 16 + idx] = (_Float16)v;
}

// ---------------------------------------------------------------------------
// WMMA GEMM with double-buffered async-to-LDS staging of B fragments.
// Block = 8 waves = 8 M-tiles x 8 N-tiles.  B chunk = 8 ntiles x 4 ktiles
// = 32KB, two LDS buffers (64KB).  Each wave always issues 8 async loads per
// chunk (ragged tails re-issue duplicate pieces) so ASYNCcnt waits are
// uniform immediates; in-order ASYNCcnt completion makes s_wait_asynccnt 8
// equivalent to "previous chunk landed".
// ---------------------------------------------------------------------------

__device__ __forceinline__ void issue_chunk(const _Float16* __restrict__ B, int nkt, int n0,
                                            int kt0, int C, int tid, unsigned ldsDst) {
    int p = tid & (64 * C - 1);   // C in {1,2,4} -> mask is power-of-two
#pragma unroll
    for (int n = 0; n < 8; ++n) {
        unsigned lds_addr = ldsDst + (unsigned)(n * 4096 + p * 16);
        unsigned long long ga =
            (unsigned long long)(const void*)(B + ((size_t)(n0 + n) * nkt + kt0) * 512)
            + (unsigned long long)p * 16ull;
        asm volatile("global_load_async_to_lds_b128 %0, %1, off"
                     :: "v"(lds_addr), "v"(ga) : "memory");
    }
}

__device__ __forceinline__ void gemm_lds_part(const _Float16* __restrict__ A,
                                              const _Float16* __restrict__ B,
                                              int nkt, int mt, int n0,
                                              int tid, int lane,
                                              unsigned ldsBase, const _Float16* ldsB,
                                              v8f* acc) {
    const v16h* Ap = (const v16h*)A + (size_t)mt * nkt * 32 + lane;
    int nch = (nkt + 3) >> 2;
    int C0 = nkt < 4 ? nkt : 4;
    issue_chunk(B, nkt, n0, 0, C0, tid, ldsBase);
    for (int i = 0; i < nch; ++i) {
        int kt0 = i * 4;
        int C = nkt - kt0;
        if (C > 4) C = 4;
        if (i + 1 < nch) {
            int C2 = nkt - (i + 1) * 4;
            if (C2 > 4) C2 = 4;
            issue_chunk(B, nkt, n0, (i + 1) * 4, C2, tid, ldsBase + ((unsigned)((i + 1) & 1)) * 32768u);
            asm volatile("s_wait_asynccnt 8" ::: "memory");
        } else {
            asm volatile("s_wait_asynccnt 0" ::: "memory");
        }
        __syncthreads();
        const v16h* Blv = (const v16h*)(ldsB + (size_t)(i & 1) * 16384);  // 32KB = 16384 halves
        for (int c = 0; c < C; ++c) {
            v16h a = Ap[(size_t)(kt0 + c) * 32];
            if (kt0 + c + 1 < nkt)
                __builtin_prefetch((const void*)(Ap + (size_t)(kt0 + c + 1) * 32), 0, 1);
#pragma unroll
            for (int n = 0; n < 8; ++n) {
                v16h b = Blv[(size_t)n * 128 + c * 32 + lane];   // n*4KB = 128 v16h
                acc[n] = __builtin_amdgcn_wmma_f32_16x16x32_f16(false, a, false, b,
                                                                (short)0, acc[n], false, false);
            }
        }
        __syncthreads();   // all readers done before buffer reuse
    }
}

__global__ __launch_bounds__(256)
void k_gemm_wmma(const _Float16* __restrict__ A1, const _Float16* __restrict__ B1, int nkt1,
                 const _Float16* __restrict__ A2, const _Float16* __restrict__ B2, int nkt2,
                 float* __restrict__ out, const float* __restrict__ residual,
                 int nMt, int nMg, int nNg, int N, int nBatch,
                 size_t aStride1, size_t aStride2, size_t oStride) {
    __shared__ _Float16 ldsB[2 * 8 * 4 * 512];   // 2 buffers x 8 ntiles x 4 ktiles = 64KB
    int tid = threadIdx.x;
    int w = tid >> 5, lane = tid & 31;
    int perB = nMg * nNg;
    int blk = blockIdx.x;
    int bt = blk / perB;
    if (bt >= nBatch) return;
    int r = blk % perB;
    int mg = r / nNg, ng = r % nNg;
    int mt = mg * 8 + w;
    if (mt >= nMt) mt = nMt - 1;   // clamp: duplicate work, identical stores
    int n0 = ng * 8;

    unsigned ldsBase = (unsigned)(unsigned long long)(void*)&ldsB[0];

    v8f acc[8];
#pragma unroll
    for (int n = 0; n < 8; ++n) acc[n] = (v8f){};

    gemm_lds_part(A1 + (size_t)bt * aStride1, B1, nkt1, mt, n0, tid, lane, ldsBase, ldsB, acc);
    if (nkt2 > 0)
        gemm_lds_part(A2 + (size_t)bt * aStride2, B2, nkt2, mt, n0, tid, lane, ldsBase, ldsB, acc);

    int colBase = n0 * 16 + (lane & 15);
    int rowBase = mt * 16 + ((lane >> 4) << 3);
#pragma unroll
    for (int n = 0; n < 8; ++n) {
        int col = colBase + n * 16;
#pragma unroll
        for (int i = 0; i < 8; ++i) {
            size_t o = (size_t)bt * oStride + (size_t)(rowBase + i) * N + col;
            float v = acc[n][i];
            if (residual) v += residual[o];
            out[o] = v;
        }
    }
}

// ---------------------------------------------------------------------------
// attention: per row r (160): scores over 512 bank entries, softmax, attn @ V
// ---------------------------------------------------------------------------

__global__ __launch_bounds__(256)
void k_attention(const float* __restrict__ q, const float* __restrict__ Kb_,
                 const float* __restrict__ Vb_, float* __restrict__ out) {
    int r = blockIdx.x;            // 0..159
    int b = r / NWAY;
    int tid = threadIdx.x;         // 256 threads
    __shared__ float qs[FD];
    __shared__ float scb[NB];
    __shared__ float red[256];

    for (int d = tid; d < FD; d += 256) qs[d] = q[(size_t)r * FD + d];
    __syncthreads();

    const float* Kb = Kb_ + (size_t)b * NB * FD;
    for (int j = tid; j < NB; j += 256) {
        const float4* kr = (const float4*)(Kb + (size_t)j * FD);
        float s = 0.f;
        for (int d4 = 0; d4 < FD / 4; ++d4) {
            float4 kv = kr[d4];
            s += kv.x * qs[4 * d4] + kv.y * qs[4 * d4 + 1] +
                 kv.z * qs[4 * d4 + 2] + kv.w * qs[4 * d4 + 3];
        }
        scb[j] = s * 0.03125f;     // 1/sqrt(1024)
    }
    __syncthreads();

    float m = fmaxf(scb[tid], scb[tid + 256]);
    red[tid] = m;
    __syncthreads();
    for (int s2 = 128; s2 > 0; s2 >>= 1) {
        if (tid < s2) red[tid] = fmaxf(red[tid], red[tid + s2]);
        __syncthreads();
    }
    float mx = red[0];
    __syncthreads();
    float e0 = expf(scb[tid] - mx), e1 = expf(scb[tid + 256] - mx);
    scb[tid] = e0; scb[tid + 256] = e1;
    red[tid] = e0 + e1;
    __syncthreads();
    for (int s2 = 128; s2 > 0; s2 >>= 1) {
        if (tid < s2) red[tid] += red[tid + s2];
        __syncthreads();
    }
    float inv = 1.f / red[0];

    const float* Vb = Vb_ + (size_t)b * NB * FD;
    float a0 = 0.f, a1 = 0.f, a2 = 0.f, a3 = 0.f;
    for (int j = 0; j < NB; ++j) {
        float w = scb[j];
        const float* vr = Vb + (size_t)j * FD;
        a0 += w * vr[tid];
        a1 += w * vr[tid + 256];
        a2 += w * vr[tid + 512];
        a3 += w * vr[tid + 768];
    }
    float* orow = out + (size_t)r * FD;
    orow[tid]       = a0 * inv;
    orow[tid + 256] = a1 * inv;
    orow[tid + 512] = a2 * inv;
    orow[tid + 768] = a3 * inv;
}

// ---------------------------------------------------------------------------
// protos: copy support centers + fake (mean over nway of out2)
// ---------------------------------------------------------------------------

__global__ void k_protos(const float* __restrict__ sc, const float* __restrict__ out2,
                         float* __restrict__ protos) {
    int t = blockIdx.x * blockDim.x + threadIdx.x;
    if (t >= BS * FD) return;
    int b = t >> 10, d = t & (FD - 1);
    float f = 0.f;
#pragma unroll
    for (int i = 0; i < NWAY; ++i) {
        size_t src = (((size_t)b * NWAY + i) << 10) + d;
        protos[(((size_t)b * (NWAY + 1) + i) << 10) + d] = sc[src];
        f += out2[src];
    }
    protos[(((size_t)b * (NWAY + 1) + NWAY) << 10) + d] = f * (1.f / NWAY);
}

__global__ void k_pnorm(const float* __restrict__ protos, float* __restrict__ pinv) {
    int gw = (blockIdx.x * blockDim.x + threadIdx.x) >> 5;
    int lane = threadIdx.x & 31;
    if (gw >= BS * (NWAY + 1)) return;
    const float* p = protos + (size_t)gw * FD;
    float s = 0.f;
    for (int d = lane; d < FD; d += 32) { float v = p[d]; s += v * v; }
    for (int o = 16; o > 0; o >>= 1) s += __shfl_xor(s, o, 32);
    if (lane == 0) pinv[gw] = 1.f / sqrtf(s);
}

__global__ void k_logits(const float* __restrict__ qf, const float* __restrict__ protos,
                         const float* __restrict__ pinv, const float* __restrict__ temp,
                         float* __restrict__ out) {
    int gw = (blockIdx.x * blockDim.x + threadIdx.x) >> 5;
    int lane = threadIdx.x & 31;
    if (gw >= BS * NQ) return;
    int b = gw / NQ, qi = gw % NQ;
    const float* q = qf + ((size_t)b * NQ + qi) * FD;
    float s = 0.f;
    for (int d = lane; d < FD; d += 32) { float v = q[d]; s += v * v; }
    for (int o = 16; o > 0; o >>= 1) s += __shfl_xor(s, o, 32);
    float qinv = 1.f / sqrtf(s);
    float tp = temp[0];
#pragma unroll
    for (int c = 0; c < NWAY + 1; ++c) {
        const float* p = protos + ((size_t)b * (NWAY + 1) + c) * FD;
        float d = 0.f;
        for (int i = lane; i < FD; i += 32) d += q[i] * p[i];
        for (int o = 16; o > 0; o >>= 1) d += __shfl_xor(d, o, 32);
        if (lane == 0)
            out[((size_t)b * NQ + qi) * (NWAY + 1) + c] = tp * d * qinv * pinv[b * (NWAY + 1) + c];
    }
}

// ---------------------------------------------------------------------------
// launch
// ---------------------------------------------------------------------------

extern "C" void kernel_launch(void* const* d_in, const int* in_sizes, int n_in,
                              void* d_out, int out_size, void* d_ws, size_t ws_size,
                              hipStream_t stream) {
    const float* sc   = (const float*)d_in[0];
    const float* bw   = (const float*)d_in[1];
    const float* ss   = (const float*)d_in[2];
    const float* bsm  = (const float*)d_in[3];
    const float* qf   = (const float*)d_in[4];
    const float* Wm1  = (const float*)d_in[5];
    const float* bm1  = (const float*)d_in[6];
    const float* Wm2  = (const float*)d_in[7];
    const float* bm2  = (const float*)d_in[8];
    const float* Wvis = (const float*)d_in[9];
    const float* bvis = (const float*)d_in[10];
    const float* Wsem = (const float*)d_in[11];
    const float* bsem = (const float*)d_in[12];
    const float* Wq   = (const float*)d_in[13];
    const float* Wk   = (const float*)d_in[14];
    const float* Wv   = (const float*)d_in[15];
    const float* Wqs  = (const float*)d_in[16];
    const float* Wks  = (const float*)d_in[17];
    const float* Wfc  = (const float*)d_in[18];
    const float* temp = (const float*)d_in[19];
    float* outp = (float*)d_out;

    char* ws = (char*)d_ws;
    size_t off = 0;
    auto alloc = [&](size_t bytes) { size_t o = off; off = (off + bytes + 255) & ~(size_t)255; return o; };

    float* h1      = (float*)(ws + alloc((size_t)BTOT * SEMD * 4));
    float* smap    = (float*)(ws + alloc((size_t)BTOT * SEMD * 4));
    float* avg     = (float*)(ws + alloc((size_t)BS * (FD + SEMD) * 4));
    float* gvis    = (float*)(ws + alloc((size_t)BS * FD * 4));
    float* gsem    = (float*)(ws + alloc((size_t)BS * SEMD * 4));
    _Float16* A_p   = (_Float16*)(ws + alloc((size_t)BS * NB * FD * 2));
    _Float16* As_p  = (_Float16*)(ws + alloc((size_t)BS * NB * SEMP * 2));
    _Float16* Wk_p  = (_Float16*)(ws + alloc((size_t)FD * FD * 2));
    _Float16* Wv_p  = (_Float16*)(ws + alloc((size_t)FD * FD * 2));
    _Float16* Wq_p  = (_Float16*)(ws + alloc((size_t)FD * FD * 2));
    _Float16* Wfc_p = (_Float16*)(ws + alloc((size_t)FD * FD * 2));
    _Float16* Wks_p = (_Float16*)(ws + alloc((size_t)SEMP * FD * 2));
    _Float16* Wqs_p = (_Float16*)(ws + alloc((size_t)SEMP * FD * 2));
    _Float16* sc_p  = (_Float16*)(ws + alloc((size_t)BTOT * FD * 2));
    _Float16* sm_p  = (_Float16*)(ws + alloc((size_t)BTOT * SEMP * 2));
    _Float16* ao_p  = (_Float16*)(ws + alloc((size_t)BTOT * FD * 2));
    float* Kbuf    = (float*)(ws + alloc((size_t)BS * NB * FD * 4));
    float* Vbuf    = (float*)(ws + alloc((size_t)BS * NB * FD * 4));
    float* qbuf    = (float*)(ws + alloc((size_t)BTOT * FD * 4));
    float* aobuf   = (float*)(ws + alloc((size_t)BTOT * FD * 4));
    float* out2    = (float*)(ws + alloc((size_t)BTOT * FD * 4));
    float* protos  = (float*)(ws + alloc((size_t)BS * (NWAY + 1) * FD * 4));
    float* pinv    = (float*)(ws + alloc((size_t)BS * (NWAY + 1) * 4));
    (void)ws_size; (void)in_sizes; (void)n_in; (void)out_size;

    // 1) semantic MLP
    k_mlp1<<<(BTOT * SEMD + 255) / 256, 256, 0, stream>>>(ss, Wm1, bm1, h1);
    k_mlp2<<<(BTOT * SEMD + 255) / 256, 256, 0, stream>>>(h1, Wm2, bm2, smap);

    // 2) task gates (depend only on bs)
    k_avg<<<(BS * (FD + SEMD) + 255) / 256, 256, 0, stream>>>(bw, bsm, avg);
    k_gates<<<(BS * (FD + SEMD) + 255) / 256, 256, 0, stream>>>(avg, Wvis, bvis, Wsem, bsem, gvis, gsem);

    // 3) pack gated bank + weights into WMMA fragment layout (f16)
    k_pack_a<<<dim3((NB * FD + 255) / 256, BS), 256, 0, stream>>>(
        bw, gvis, A_p, NB, FD, FD, (size_t)NB * FD, FD, (size_t)NB * FD);
    k_pack_a<<<dim3((NB * SEMP + 255) / 256, BS), 256, 0, stream>>>(
        bsm, gsem, As_p, NB, SEMD, SEMP, (size_t)NB * SEMD, SEMD, (size_t)NB * SEMP);
    k_pack_b<<<(FD * FD + 255) / 256, 256, 0, stream>>>(Wk, Wk_p, FD, FD, FD);
    k_pack_b<<<(FD * FD + 255) / 256, 256, 0, stream>>>(Wv, Wv_p, FD, FD, FD);
    k_pack_b<<<(FD * FD + 255) / 256, 256, 0, stream>>>(Wq, Wq_p, FD, FD, FD);
    k_pack_b<<<(FD * FD + 255) / 256, 256, 0, stream>>>(Wfc, Wfc_p, FD, FD, FD);
    k_pack_b<<<(SEMP * FD + 255) / 256, 256, 0, stream>>>(Wks, Wks_p, SEMD, SEMP, FD);
    k_pack_b<<<(SEMP * FD + 255) / 256, 256, 0, stream>>>(Wqs, Wqs_p, SEMD, SEMP, FD);
    k_pack_a<<<dim3((BTOT * FD + 255) / 256, 1), 256, 0, stream>>>(
        sc, (const float*)nullptr, sc_p, BTOT, FD, FD, 0, 0, 0);
    k_pack_a<<<dim3((BTOT * SEMP + 255) / 256, 1), 256, 0, stream>>>(
        smap, (const float*)nullptr, sm_p, BTOT, SEMD, SEMP, 0, 0, 0);

    // 4) dominant GEMMs: K = A@Wk + As@Wks ; V = A@Wv   (per bs, 32 batches)
    const int nMt = NB / 16, nMg = nMt / 8, nNg = FD / 128;  // 32 Mtiles, 4 Mgroups, 8 Ngroups
    int kvBlocks = BS * nMg * nNg;                           // 1024 blocks
    k_gemm_wmma<<<kvBlocks, 256, 0, stream>>>(
        A_p, Wk_p, FD / 32, As_p, Wks_p, SEMP / 32, Kbuf, nullptr,
        nMt, nMg, nNg, FD, BS, (size_t)NB * FD, (size_t)NB * SEMP, (size_t)NB * FD);
    k_gemm_wmma<<<kvBlocks, 256, 0, stream>>>(
        A_p, Wv_p, FD / 32, nullptr, nullptr, 0, Vbuf, nullptr,
        nMt, nMg, nNg, FD, BS, (size_t)NB * FD, 0, (size_t)NB * FD);

    // 5) q = sc@Wq + smap@Wqs   (M=160 -> 10 Mtiles -> 2 Mgroups with clamp)
    int qMt = BTOT / 16, qMg = (qMt + 7) / 8;
    int qBlocks = qMg * nNg;                                 // 16 blocks
    k_gemm_wmma<<<qBlocks, 256, 0, stream>>>(
        sc_p, Wq_p, FD / 32, sm_p, Wqs_p, SEMP / 32, qbuf, nullptr,
        qMt, qMg, nNg, FD, 1, 0, 0, 0);

    // 6) attention + softmax + attn@V
    k_attention<<<BTOT, 256, 0, stream>>>(qbuf, Kbuf, Vbuf, aobuf);

    // 7) out2 = attn_out @ Wfc + sc   (WMMA + f32 residual)
    k_pack_a<<<dim3((BTOT * FD + 255) / 256, 1), 256, 0, stream>>>(
        aobuf, (const float*)nullptr, ao_p, BTOT, FD, FD, 0, 0, 0);
    k_gemm_wmma<<<qBlocks, 256, 0, stream>>>(
        ao_p, Wfc_p, FD / 32, nullptr, nullptr, 0, out2, sc,
        qMt, qMg, nNg, FD, 1, 0, 0, 0);

    // 8) prototypes + cosine logits
    k_protos<<<(BS * FD + 255) / 256, 256, 0, stream>>>(sc, out2, protos);
    k_pnorm<<<(BS * (NWAY + 1) * 32 + 255) / 256, 256, 0, stream>>>(protos, pinv);
    k_logits<<<(BS * NQ * 32 + 255) / 256, 256, 0, stream>>>(qf, protos, pinv, temp, outp);
}
